// WindowAttention3D_14783277433305
// MI455X (gfx1250) — compile-verified, matrix operand order checked by reading, not run on hardware
//
#include <hip/hip_runtime.h>

// ---------------------------------------------------------------------------
// Fused Swin-3D window attention for gfx1250 (MI455X), wave32 + WMMA + TDM.
// One block per window (2048 blocks), 256 threads = 8 waves.
// All GEMMs use v_wmma_f32_16x16x32_f16; softmax in fp32.
// x-window and mask-window are DMA'd into LDS with tensor_load_to_lds and
// synchronized with s_wait_tensorcnt.
// ---------------------------------------------------------------------------

#define DIM     128
#define HEADS   4
#define HD      32
#define N_TOK   98
#define NPAD    112   // 7 * 16
#define KPAD    128   // padded K for AV gemm
#define NW      256
#define MT      7     // M tiles (112/16)

typedef __attribute__((ext_vector_type(16))) _Float16 v16h;
typedef __attribute__((ext_vector_type(8)))  _Float16 v8h;
typedef __attribute__((ext_vector_type(8)))  float    v8f;
typedef __attribute__((ext_vector_type(4))) unsigned int u32x4;
typedef __attribute__((ext_vector_type(8))) int          i32x8;
typedef __attribute__((ext_vector_type(4))) int          i32x4;

// Load a 16x32 f16 fragment (A layout; B loaded from B^T row-major) from LDS.
// Caller passes p = base + row*ld + kb where kb = (lane>>4)*8.
// Halves 0..7 <- p[0..8), halves 8..15 <- p[16..24).
__device__ __forceinline__ v16h frag_lds(const _Float16* p) {
    v8h lo = *(const v8h*)p;
    v8h hi = *(const v8h*)(p + 16);
    v16h f;
#pragma unroll
    for (int i = 0; i < 8; ++i) { f[i] = lo[i]; f[8 + i] = hi[i]; }
    return f;
}

// Same fragment but sourced from global fp32 (weights), converted to f16.
__device__ __forceinline__ v16h frag_gf32(const float* p) {
    float4 a0 = *(const float4*)(p + 0);
    float4 a1 = *(const float4*)(p + 4);
    float4 b0 = *(const float4*)(p + 16);
    float4 b1 = *(const float4*)(p + 20);
    v16h f;
    f[0]=(_Float16)a0.x; f[1]=(_Float16)a0.y; f[2]=(_Float16)a0.z; f[3]=(_Float16)a0.w;
    f[4]=(_Float16)a1.x; f[5]=(_Float16)a1.y; f[6]=(_Float16)a1.z; f[7]=(_Float16)a1.w;
    f[8]=(_Float16)b0.x; f[9]=(_Float16)b0.y; f[10]=(_Float16)b0.z; f[11]=(_Float16)b0.w;
    f[12]=(_Float16)b1.x; f[13]=(_Float16)b1.y; f[14]=(_Float16)b1.z; f[15]=(_Float16)b1.w;
    return f;
}

// Issue one 2-D fp32 tile DMA: global (dim1 rows x dim0 cols, row stride
// stride0 elements) -> LDS byte offset lds_off.  D# layout per CDNA5 ISA
// ch. 8 (group0: count/lds/global addr/type; group1: data_size, dims, tiles,
// strides; groups 2/3 zero = 2-D tensor).
__device__ __forceinline__ void tdm_load_2d_f32(unsigned lds_off, const void* gptr,
                                                unsigned dim0, unsigned dim1,
                                                unsigned tile0, unsigned tile1,
                                                unsigned stride0) {
    unsigned long long ga = (unsigned long long)(uintptr_t)gptr;
    u32x4 g0;
    g0[0] = 1u;                                              // count=1, user D#
    g0[1] = lds_off;                                         // lds_addr (bytes)
    g0[2] = (unsigned)(ga & 0xFFFFFFFFu);                    // global_addr[31:0]
    g0[3] = (unsigned)((ga >> 32) & 0x01FFFFFFu) | (2u << 30); // addr[56:32] | type=2
    i32x8 g1;
    g1[0] = (int)(2u << 16);                                 // data_size = 4 bytes
    g1[1] = (int)((dim0 & 0xFFFFu) << 16);                   // tensor_dim0[15:0]
    g1[2] = (int)((dim0 >> 16) | ((dim1 & 0xFFFFu) << 16));  // dim0 hi | dim1 lo
    g1[3] = (int)((dim1 >> 16) | (tile0 << 16));             // dim1 hi | tile_dim0
    g1[4] = (int)(tile1 & 0xFFFFu);                          // tile_dim1 (tile_dim2=0)
    g1[5] = (int)stride0;                                    // tensor_dim0_stride lo
    g1[6] = 0;                                               // stride hi | dim1_stride lo
    g1[7] = 0;
    i32x4 z4 = {0, 0, 0, 0};
    i32x8 z8 = {0, 0, 0, 0, 0, 0, 0, 0};
    __builtin_amdgcn_tensor_load_to_lds(g0, g1, z4, z4, z8, 0);
}

extern "C" __global__ __launch_bounds__(256, 1)
void win_attn3d_wmma(const float* __restrict__ x,
                     const float* __restrict__ mask,
                     const float* __restrict__ qkv_w,
                     const float* __restrict__ qkv_b,
                     const float* __restrict__ proj_w,
                     const float* __restrict__ proj_b,
                     const float* __restrict__ bias_table,
                     const int*   __restrict__ rel_index,
                     float* __restrict__ out)
{
    extern __shared__ __align__(16) char smem_raw[];
    _Float16* XH   = (_Float16*)smem_raw;            // 112x128 f16   x window
    _Float16* QH   = XH   + NPAD * KPAD;             // 4 x 112x32    q (scaled)
    _Float16* KH   = QH   + HEADS * NPAD * HD;       // 4 x 112x32    k (== B^T)
    _Float16* OUTH = KH   + HEADS * NPAD * HD;       // 112x128       attn @ v
    _Float16* AH   = OUTH + NPAD * KPAD;             // 112x128       softmax(A) as f16
    _Float16* VT   = AH   + NPAD * KPAD;             // 4 x 32x128    v^T, zero-padded
    float*    ATTN = (float*)(VT + HEADS * HD * KPAD); // 112x112 f32 scores
    float*    XF   = ATTN;                           // alias: 98x128 f32 x staging
                                                     // (dead before ATTN first use)
    float*    MASKL= ATTN + NPAD * NPAD;             // 98x98 f32 mask window

    const int b    = blockIdx.x;
    const int tid  = threadIdx.x;
    const int lane = tid & 31;
    const int wave = tid >> 5;
    const int fr   = lane & 15;            // fragment row (A) / col (B/C)
    const int kb   = (lane >> 4) << 3;     // fragment K chunk base (0 or 8)
    const int crow = (lane >> 4) << 3;     // C-fragment row base (0 or 8)

    // ---- wave 0: kick off TDM DMAs (x window fp32, mask window fp32) ----
    if (wave == 0) {
        tdm_load_2d_f32((unsigned)(uintptr_t)(void*)XF,
                        x + (size_t)b * N_TOK * DIM,
                        DIM, N_TOK, DIM, N_TOK, DIM);
        tdm_load_2d_f32((unsigned)(uintptr_t)(void*)MASKL,
                        mask + (size_t)(b % NW) * N_TOK * N_TOK,
                        N_TOK, N_TOK, N_TOK, N_TOK, N_TOK);
    }

    // ---- zero-init AH + VT (contiguous region; pads must stay 0) ----
    {
        uint4* z = (uint4*)AH;
        const int nz = (NPAD * KPAD + HEADS * HD * KPAD) * (int)sizeof(_Float16) / 16;
        for (int i = tid; i < nz; i += 256) z[i] = make_uint4(0, 0, 0, 0);
    }

    // ---- wait for x DMA (first tensor op), then convert fp32 -> f16 ----
    if (wave == 0) __builtin_amdgcn_s_wait_tensorcnt(1);
    __syncthreads();
    for (int i = tid; i < NPAD * (DIM / 4); i += 256) {
        int r = i / (DIM / 4);
        int c = (i % (DIM / 4)) * 4;
        float4 v4 = make_float4(0.f, 0.f, 0.f, 0.f);
        if (r < N_TOK) v4 = *(const float4*)(XF + r * DIM + c);
        _Float16* d = XH + r * KPAD + c;
        d[0] = (_Float16)v4.x; d[1] = (_Float16)v4.y;
        d[2] = (_Float16)v4.z; d[3] = (_Float16)v4.w;
    }
    __syncthreads();

    // ================= Phase 1: qkv = x @ qkv_w^T + qkv_b =================
    // M=112 (7 tiles), N=384 (24 tiles), K=128 (4 wmma steps).
    const float qscale = 0.17677669529663687f;  // 32^-0.5
    for (int t = wave; t < MT * 24; t += 8) {
        int mt = t / 24, nt = t % 24;
        v8f c = {};
#pragma unroll
        for (int ks = 0; ks < 4; ++ks) {
            v16h a  = frag_lds(XH + (mt * 16 + fr) * KPAD + ks * 32 + kb);
            v16h bm = frag_gf32(qkv_w + (size_t)(nt * 16 + fr) * DIM + ks * 32 + kb);
            c = __builtin_amdgcn_wmma_f32_16x16x32_f16(false, a, false, bm,
                                                       (short)0, c, false, false);
        }
        int col  = nt * 16 + fr;              // 0..383
        float bi = qkv_b[col];
        int kind = col >> 7;                  // 0=q, 1=k, 2=v
        int h    = (col & 127) >> 5;
        int d    = col & 31;
#pragma unroll
        for (int v = 0; v < 8; ++v) {
            int m = mt * 16 + v + crow;
            if (m < N_TOK) {
                float val = c[v] + bi;
                if (kind == 0)      QH[(h * NPAD + m) * HD + d] = (_Float16)(val * qscale);
                else if (kind == 1) KH[(h * NPAD + m) * HD + d] = (_Float16)val;
                else                VT[(h * HD + d) * KPAD + m] = (_Float16)val;
            }
        }
    }
    // make sure the mask DMA has landed before first softmax
    if (wave == 0) __builtin_amdgcn_s_wait_tensorcnt(0);
    __syncthreads();

    // ================= Phase 2-4: per-head attention =================
    for (int h = 0; h < HEADS; ++h) {
        const _Float16* Q  = QH + h * NPAD * HD;
        const _Float16* K  = KH + h * NPAD * HD;   // row-major k == B^T of K^T
        const _Float16* Vh = VT + h * HD * KPAD;   // v^T == B^T of V

        // ---- scores = q @ k^T  (K=32 -> single wmma per tile) ----
        for (int t = wave; t < MT * MT; t += 8) {
            int mt = t / MT, nt = t % MT;
            v16h a  = frag_lds(Q + (mt * 16 + fr) * HD + kb);
            v16h bm = frag_lds(K + (nt * 16 + fr) * HD + kb);
            v8f c = {};
            c = __builtin_amdgcn_wmma_f32_16x16x32_f16(false, a, false, bm,
                                                       (short)0, c, false, false);
#pragma unroll
            for (int v = 0; v < 8; ++v)
                ATTN[(mt * 16 + v + crow) * NPAD + nt * 16 + fr] = c[v];
        }
        __syncthreads();

        // ---- +bias +mask (from LDS), softmax (fp32), emit f16 A-matrix ----
        if (tid < N_TOK) {
            int r = tid;
            float*       arow = ATTN + r * NPAD;
            const int*   rrow = rel_index + r * N_TOK;
            const float* mrow = MASKL + r * N_TOK;
            float mx = -3.0e38f;
            for (int n = 0; n < N_TOK; ++n) {
                float val = arow[n] + bias_table[rrow[n] * HEADS + h] + mrow[n];
                arow[n] = val;
                mx = fmaxf(mx, val);
            }
            float s = 0.f;
            for (int n = 0; n < N_TOK; ++n) {
                float e = __expf(arow[n] - mx);
                arow[n] = e;
                s += e;
            }
            float inv = 1.0f / s;
            _Float16* ar = AH + r * KPAD;
            for (int n = 0; n < N_TOK; ++n) ar[n] = (_Float16)(arow[n] * inv);
            for (int n = N_TOK; n < NPAD; ++n) ar[n] = (_Float16)0.f;  // kill pad cols
        }
        __syncthreads();

        // ---- out_h = softmax(A) @ v   (M=112, N=32, K=128 padded) ----
        for (int t = wave; t < MT * 2; t += 8) {
            int mt = t >> 1, dt = t & 1;
            v8f c = {};
#pragma unroll
            for (int ks = 0; ks < 4; ++ks) {
                v16h a  = frag_lds(AH + (mt * 16 + fr) * KPAD + ks * 32 + kb);
                v16h bm = frag_lds(Vh + (dt * 16 + fr) * KPAD + ks * 32 + kb);
                c = __builtin_amdgcn_wmma_f32_16x16x32_f16(false, a, false, bm,
                                                           (short)0, c, false, false);
            }
#pragma unroll
            for (int v = 0; v < 8; ++v)
                OUTH[(mt * 16 + v + crow) * KPAD + h * HD + dt * 16 + fr] = (_Float16)c[v];
        }
        __syncthreads();
    }

    // ================= Phase 5: final proj + bias, store fp32 =================
    {
        float* ob = out + (size_t)b * N_TOK * DIM;
        for (int t = wave; t < MT * 8; t += 8) {
            int mt = t >> 3, nt = t & 7;   // nt == wave
            v8f c = {};
#pragma unroll
            for (int ks = 0; ks < 4; ++ks) {
                v16h a  = frag_lds(OUTH + (mt * 16 + fr) * KPAD + ks * 32 + kb);
                v16h bm = frag_gf32(proj_w + (size_t)(nt * 16 + fr) * DIM + ks * 32 + kb);
                c = __builtin_amdgcn_wmma_f32_16x16x32_f16(false, a, false, bm,
                                                           (short)0, c, false, false);
            }
            int col  = nt * 16 + fr;
            float pb = proj_b[col];
#pragma unroll
            for (int v = 0; v < 8; ++v) {
                int m = mt * 16 + v + crow;
                if (m < N_TOK) ob[(size_t)m * DIM + col] = c[v] + pb;
            }
        }
    }
}

extern "C" void kernel_launch(void* const* d_in, const int* in_sizes, int n_in,
                              void* d_out, int out_size, void* d_ws, size_t ws_size,
                              hipStream_t stream) {
    (void)in_sizes; (void)n_in; (void)out_size; (void)d_ws; (void)ws_size;
    const float* x          = (const float*)d_in[0];
    const float* mask       = (const float*)d_in[1];
    const float* qkv_w      = (const float*)d_in[2];
    const float* qkv_b      = (const float*)d_in[3];
    const float* proj_w     = (const float*)d_in[4];
    const float* proj_b     = (const float*)d_in[5];
    const float* bias_table = (const float*)d_in[6];
    const int*   rel_index  = (const int*)d_in[7];
    float* out = (float*)d_out;

    // f16: XH + QH + KH + OUTH + AH + VT ; f32: ATTN(/XF alias) + MASKL
    const size_t smem =
        (size_t)(NPAD * KPAD + 2 * HEADS * NPAD * HD + 2 * NPAD * KPAD +
                 HEADS * HD * KPAD) * sizeof(_Float16) +
        (size_t)(NPAD * NPAD + N_TOK * N_TOK) * sizeof(float);  // 264,720 B < 320 KB

    (void)hipFuncSetAttribute((const void*)win_attn3d_wmma,
                              hipFuncAttributeMaxDynamicSharedMemorySize, (int)smem);

    win_attn3d_wmma<<<2048, 256, smem, stream>>>(
        x, mask, qkv_w, qkv_b, proj_w, proj_b, bias_table, rel_index, out);
}
